// GraphNN_4836133175863
// MI455X (gfx1250) — compile-verified, matrix operand order checked by reading, not run on hardware
//
#include <hip/hip_runtime.h>

#define DD 256
#define HH 512
#define BN_EPS 1e-5f

typedef __attribute__((ext_vector_type(16))) __bf16 v16bf;
typedef __attribute__((ext_vector_type(8)))  __bf16 v8bf;
typedef __attribute__((ext_vector_type(8)))  float  v8f;
typedef __attribute__((ext_vector_type(4)))  float  v4f;
typedef __attribute__((ext_vector_type(4)))  __bf16 v4bf;

// ---------------- degree / norm ----------------

__global__ void k_fill_ones(float* deg, int n) {
    int i = blockIdx.x * blockDim.x + threadIdx.x;
    if (i < n) deg[i] = 1.0f;                      // self-loop contribution
}

__global__ void k_deg_count(const int* __restrict__ dst, float* deg, int E) {
    int e = blockIdx.x * blockDim.x + threadIdx.x;
    if (e < E) atomicAdd(&deg[dst[e]], 1.0f);
}

__global__ void k_rsqrt_inplace(float* d, int n) {
    int i = blockIdx.x * blockDim.x + threadIdx.x;
    if (i < n) d[i] = rsqrtf(d[i]);                // deg >= 1 always
}

// ---------------- conversions ----------------

__global__ void k_cvt_f32_bf16(const float* __restrict__ x, __bf16* __restrict__ y, int n4) {
    // n4 = n/4, n divisible by 4
    for (int i = blockIdx.x * blockDim.x + threadIdx.x; i < n4; i += gridDim.x * blockDim.x) {
        v4f a = ((const v4f*)x)[i];
        ((v4bf*)y)[i] = __builtin_convertvector(a, v4bf);
    }
}

__global__ void k_transpose_bf16(const float* __restrict__ W, __bf16* __restrict__ Wt,
                                 int K, int Nc) {
    // W: K x Nc row-major -> Wt: Nc x K row-major (bf16)
    int idx = blockIdx.x * blockDim.x + threadIdx.x;
    if (idx < K * Nc) {
        int k = idx / Nc, n = idx % Nc;
        Wt[(size_t)n * K + k] = (__bf16)W[idx];
    }
}

// ---------------- WMMA GEMM: C[M x Nc] = A[M x K] * Bt[Nc x K]^T ----------------
// A, Bt bf16 row-major; K % 32 == 0, Nc % 64 == 0.

__device__ __forceinline__ v16bf load_frag(const __bf16* __restrict__ base, int stride,
                                           int row, int rowMax, int kb, int lane) {
    int r  = lane & 15;
    int hi = lane >> 4;
    int rr = row + r;
    rr = rr < rowMax ? rr : rowMax - 1;            // branch-free clamp, EXEC stays full
    const __bf16* p = base + (size_t)rr * stride + kb + hi * 8;
    v8bf lo  = *(const v8bf*)p;                    // K = kb + hi*8 .. +7      -> v0..v3
    v8bf hiv = *(const v8bf*)(p + 16);             // K = kb + 16 + hi*8 .. +7 -> v4..v7
    return __builtin_shufflevector(lo, hiv, 0,1,2,3,4,5,6,7,8,9,10,11,12,13,14,15);
}

__global__ void __launch_bounds__(32)
k_gemm_bf16_wmma(const __bf16* __restrict__ A, const __bf16* __restrict__ Bt,
                 float* __restrict__ C, int M, int K, int Nc) {
    int lane = threadIdx.x;
    int m0 = blockIdx.x * 32;
    int n0 = blockIdx.y * 64;

    v8f acc[2][4] = {};
    for (int kb = 0; kb < K; kb += 32) {
        v16bf a0 = load_frag(A, K, m0,      M,  kb, lane);
        v16bf a1 = load_frag(A, K, m0 + 16, M,  kb, lane);
#pragma unroll
        for (int j = 0; j < 4; ++j) {
            v16bf b = load_frag(Bt, K, n0 + j * 16, Nc, kb, lane);
            acc[0][j] = __builtin_amdgcn_wmma_f32_16x16x32_bf16(
                false, a0, false, b, (short)0, acc[0][j], false, false);
            acc[1][j] = __builtin_amdgcn_wmma_f32_16x16x32_bf16(
                false, a1, false, b, (short)0, acc[1][j], false, false);
        }
    }

    // C/D layout: vgpr r, lanes 0-15 -> M = r, N = lane; lanes 16-31 -> M = 8+r, N = lane-16
    int col = lane & 15;
    int hi  = lane >> 4;
#pragma unroll
    for (int i = 0; i < 2; ++i) {
#pragma unroll
        for (int j = 0; j < 4; ++j) {
            int n = n0 + j * 16 + col;
#pragma unroll
            for (int r = 0; r < 8; ++r) {
                int m = m0 + i * 16 + hi * 8 + r;
                if (m < M) C[(size_t)m * Nc + n] = acc[i][j][r];
            }
        }
    }
}

// ---------------- aggregation ----------------

// out[i,:] = bias + xw[i,:] * dinv[i]^2   (self-loop message + bias, initializes accumulator)
__global__ void k_self_init(float* __restrict__ out, const float* __restrict__ xw,
                            const float* __restrict__ dinv, const float* __restrict__ bias,
                            int F) {
    int row = blockIdx.x;
    float d = dinv[row];
    float d2 = d * d;
    const float* xr = xw + (size_t)row * F;
    float* orow = out + (size_t)row * F;
    for (int f = threadIdx.x; f < F; f += blockDim.x)
        orow[f] = bias[f] + xr[f] * d2;
}

// out[dst,:] += xw[src,:] * dinv[src]*dinv[dst]   (one block per edge)
__global__ void k_edge_agg(float* __restrict__ out, const float* __restrict__ xw,
                           const float* __restrict__ dinv,
                           const int* __restrict__ src, const int* __restrict__ dst,
                           int F) {
    int e = blockIdx.x;
    int s = src[e], d = dst[e];
    float norm = dinv[s] * dinv[d];
    const float* xs = xw + (size_t)s * F;
    float* od = out + (size_t)d * F;
    for (int f = threadIdx.x; f < F; f += blockDim.x)
        atomicAdd(&od[f], xs[f] * norm);
}

// ---------------- batch norm ----------------

__global__ void k_zero(float* p, int n) {
    int i = blockIdx.x * blockDim.x + threadIdx.x;
    if (i < n) p[i] = 0.0f;
}

__global__ void k_bn_stats(const float* __restrict__ x, float* __restrict__ sums, int n_rows) {
    int c = threadIdx.x;                            // blockDim.x == DD
    float s = 0.0f, s2 = 0.0f;
    for (int r = blockIdx.x; r < n_rows; r += gridDim.x) {
        float v = x[(size_t)r * DD + c];
        s += v; s2 += v * v;
    }
    atomicAdd(&sums[c], s);
    atomicAdd(&sums[DD + c], s2);
}

__global__ void k_bn_apply(float* __restrict__ out, const float* __restrict__ sums,
                           const float* __restrict__ gamma, const float* __restrict__ beta,
                           int n_rows) {
    int idx = blockIdx.x * blockDim.x + threadIdx.x;
    if (idx >= n_rows * DD) return;
    int c = idx & (DD - 1);
    float inv_n = 1.0f / (float)n_rows;
    float mean = sums[c] * inv_n;
    float var  = sums[DD + c] * inv_n - mean * mean;
    float rstd = rsqrtf(var + BN_EPS);
    out[idx] = gamma[c] * (out[idx] - mean) * rstd + beta[c];
}

// ---------------- host ----------------

extern "C" void kernel_launch(void* const* d_in, const int* in_sizes, int n_in,
                              void* d_out, int out_size, void* d_ws, size_t ws_size,
                              hipStream_t stream) {
    const int*   edge_index = (const int*)d_in[0];
    const float* emb   = (const float*)d_in[1];
    const float* W1    = (const float*)d_in[2];
    const float* b1    = (const float*)d_in[3];
    const float* W2    = (const float*)d_in[4];
    const float* b2    = (const float*)d_in[5];
    const float* gamma = (const float*)d_in[6];
    const float* beta  = (const float*)d_in[7];

    const int E = in_sizes[0] / 2;
    const int N = in_sizes[1] / DD;
    const int* src = edge_index;
    const int* dst = edge_index + E;
    float* out = (float*)d_out;

    auto align256 = [](size_t x) { return (x + 255) & ~(size_t)255; };
    char* ws = (char*)d_ws;
    size_t off = 0;
    float*  dinv = (float*)(ws + off);  off = align256(off + (size_t)N * 4);
    __bf16* xbf  = (__bf16*)(ws + off); off = align256(off + (size_t)N * DD * 2);
    __bf16* w1t  = (__bf16*)(ws + off); off = align256(off + (size_t)DD * HH * 2);
    __bf16* w2t  = (__bf16*)(ws + off); off = align256(off + (size_t)DD * HH * 2);
    char*   bufA = ws + off;            off = align256(off + (size_t)N * HH * 4);
    float*  h1   = (float*)(ws + off);  off = align256(off + (size_t)N * HH * 4);
    float*  sums = (float*)(ws + off);  off = align256(off + (size_t)2 * DD * 4);

    float*  xw1  = (float*)bufA;                          // N x HH f32 (layer-1 GEMM out)
    __bf16* h1bf = (__bf16*)bufA;                         // reuse: N x HH bf16
    float*  xw2  = (float*)(bufA + (size_t)N * HH * 2);   // reuse: N x DD f32

    const int T = 256;

    // degree -> dinv
    k_fill_ones<<<(N + T - 1) / T, T, 0, stream>>>(dinv, N);
    k_deg_count<<<(E + T - 1) / T, T, 0, stream>>>(dst, dinv, E);
    k_rsqrt_inplace<<<(N + T - 1) / T, T, 0, stream>>>(dinv, N);

    // precision staging
    k_cvt_f32_bf16<<<2048, T, 0, stream>>>(emb, xbf, N * DD / 4);
    k_transpose_bf16<<<(DD * HH + T - 1) / T, T, 0, stream>>>(W1, w1t, DD, HH);
    k_transpose_bf16<<<(DD * HH + T - 1) / T, T, 0, stream>>>(W2, w2t, HH, DD);

    // layer 1: xw1 = x @ W1 ; h1 = bias + aggregate(xw1)
    {
        dim3 grid((N + 31) / 32, HH / 64);
        k_gemm_bf16_wmma<<<grid, 32, 0, stream>>>(xbf, w1t, xw1, N, DD, HH);
    }
    k_self_init<<<N, T, 0, stream>>>(h1, xw1, dinv, b1, HH);
    k_edge_agg<<<E, T, 0, stream>>>(h1, xw1, dinv, src, dst, HH);

    // layer 2: xw2 = h1 @ W2 ; out = bias + aggregate(xw2)
    k_cvt_f32_bf16<<<2048, T, 0, stream>>>(h1, h1bf, N * HH / 4);
    {
        dim3 grid((N + 31) / 32, DD / 64);
        k_gemm_bf16_wmma<<<grid, 32, 0, stream>>>(h1bf, w2t, xw2, N, HH, DD);
    }
    k_self_init<<<N, T, 0, stream>>>(out, xw2, dinv, b2, DD);
    k_edge_agg<<<E, T, 0, stream>>>(out, xw2, dinv, src, dst, DD);

    // batch norm over node dim, in place on d_out
    k_zero<<<1, 2 * DD, 0, stream>>>(sums, 2 * DD);
    k_bn_stats<<<512, DD, 0, stream>>>(out, sums, N);
    k_bn_apply<<<(N * DD + T - 1) / T, T, 0, stream>>>(out, sums, gamma, beta, N);
}
